// VanillaMPN_60627758350346
// MI455X (gfx1250) — matrix operand order repaired and checked by reading, not compiled, not run on hardware
//
#include <hip/hip_runtime.h>

// ---------------------------------------------------------------------------
// MI455X (gfx1250, wave32) fused MPN.
// All GEMMs use v_wmma_f32_16x16x32_bf16. Weights pre-packed into B-fragment
// layout (per-lane 16 contiguous bf16 -> two b128 loads). Features kept in
// bf16 to halve random-gather bytes (problem is gather-bound: ~85 GFLOP total
// vs >1 GB of scattered traffic; L2-resident at 192MB, 23.3 TB/s HBM).
// kb-outer / nt-inner with constant-indexed accumulator arrays. Weight bases
// are offset-laundered per tile-iteration: blocks LICM from hoisting (and
// spilling) B-fragment loads, while preserving pointer provenance so loads
// stay global_load_b128 (FLAT would also tick DScnt and over-serialize).
// ---------------------------------------------------------------------------

typedef __attribute__((ext_vector_type(8)))  float   v8f;
typedef __attribute__((ext_vector_type(8)))  __bf16  v8bf;
typedef __attribute__((ext_vector_type(16))) __bf16  v16bf;

static constexpr int NNODE = 20000;
static constexpr int NEDGE = 100000;
static constexpr int FD    = 128;   // node/edge feature dim

__device__ __forceinline__ v16bf cat8(v8bf lo, v8bf hi) {
  return __builtin_shufflevector(lo, hi, 0,1,2,3,4,5,6,7,8,9,10,11,12,13,14,15);
}

// Opaque zero offset (SGPR): defeats loop-invariant hoisting of the loads
// that feed WMMA while keeping the base pointer's global address space.
__device__ __forceinline__ const __bf16* launder(const __bf16* p) {
  unsigned int z = 0;
  asm volatile("" : "+s"(z));
  return p + z;
}

// A-fragment (16x32 bf16, MxK): lane l -> row l&15; K elems {k0+hi..k0+hi+7,
// k0+16+hi..k0+16+hi+7} with hi = (lane>=16) ? 8 : 0. Two b128 loads.
__device__ __forceinline__ v16bf afrag_bf16(const __bf16* row, int k0, int hi) {
  v8bf lo = *(const v8bf*)(row + k0 + hi);
  v8bf h2 = *(const v8bf*)(row + k0 + 16 + hi);
  return cat8(lo, h2);
}

// Same fragment gathered from f32 source with K bounds guard (for EIN=291 pad).
__device__ __forceinline__ v16bf afrag_f32(const float* row, int k0, int hi, int K) {
  v16bf a;
  int b0 = k0 + hi, b1 = k0 + 16 + hi;
  if ((k0 + 32 <= K) && ((((unsigned long long)row) & 31ull) == 0)) {
    v8f lo = *(const v8f*)(row + b0);
    v8f h2 = *(const v8f*)(row + b1);
#pragma unroll
    for (int e = 0; e < 8; ++e) { a[e] = (__bf16)lo[e]; a[8+e] = (__bf16)h2[e]; }
  } else {
#pragma unroll
    for (int e = 0; e < 8; ++e) {
      int ka = b0 + e, kb = b1 + e;
      a[e]   = (__bf16)((ka < K) ? row[ka] : 0.0f);
      a[8+e] = (__bf16)((kb < K) ? row[kb] : 0.0f);
    }
  }
  return a;
}

// B-fragment from pre-packed weights: fragment = 32 lanes x 16 bf16 = 1KB.
__device__ __forceinline__ v16bf bfrag(const __bf16* pack, int frag, int lane) {
  const __bf16* p = pack + ((size_t)frag << 9) + (lane << 4);
  v8bf lo = *(const v8bf*)p;
  v8bf h2 = *(const v8bf*)(p + 8);
  return cat8(lo, h2);
}

__device__ __forceinline__ v8f wmma_bf16(v16bf a, v16bf b, v8f c) {
  return __builtin_amdgcn_wmma_f32_16x16x32_bf16(false, a, false, b, (short)0, c,
                                                 false, false);
}

__device__ __forceinline__ void wait_lds() {
  asm volatile("s_wait_dscnt 0" ::: "memory");
}

__device__ __forceinline__ void atomAddF(float* p, float v) {
  unsafeAtomicAdd(p, v);   // global_atomic_add_f32
}

// ---------------------------------------------------------------------------
// Pack a row-major f32 weight [K,N] into WMMA B-fragment bf16 layout.
// frag(kb,nt): lane l holds n=nt*16+(l&15); element e holds
// k=kb*32+8*(l>=16)+(e&7)+16*(e>=8) (matches A-fragment K order).
// ---------------------------------------------------------------------------
__global__ void pack_w_kernel(const float* __restrict__ W, __bf16* __restrict__ dst,
                              int K, int N, int KB, int NT) {
  int idx = blockIdx.x * blockDim.x + threadIdx.x;
  int total = KB * NT * 512;
  if (idx >= total) return;
  int e    = idx & 15;
  int lane = (idx >> 4) & 31;
  int frag = idx >> 9;
  int kb = frag / NT, nt = frag % NT;
  int k = kb * 32 + ((lane & 16) ? 8 : 0) + (e & 7) + ((e & 8) ? 16 : 0);
  int n = nt * 16 + (lane & 15);
  float v = (k < K && n < N) ? W[(size_t)k * N + n] : 0.0f;
  dst[idx] = (__bf16)v;
}

__global__ void zero_f32_kernel(float* __restrict__ p, int n) {
  int i = blockIdx.x * blockDim.x + threadIdx.x;
  if (i < n) p[i] = 0.0f;
}

__global__ void cvt_f32_bf16_kernel(const float* __restrict__ s, __bf16* __restrict__ d, int n) {
  int i = blockIdx.x * blockDim.x + threadIdx.x;
  if (i < n) d[i] = (__bf16)s[i];
}

// ---------------------------------------------------------------------------
// Generic fused 2-layer MLP: out = act2(relu(in @ W1 + b1) @ W2 + b2).
// One wave per 16-row tile; hidden tile staged in per-wave LDS.
// ---------------------------------------------------------------------------
template <bool IN_F32, bool RELU2, int H, int N2>
__global__ void mlp2_kernel(const void* __restrict__ inv, int ldin, int Mtiles, int K,
                            const __bf16* __restrict__ W1g, const float* __restrict__ b1,
                            const __bf16* __restrict__ W2g, const float* __restrict__ b2,
                            __bf16* __restrict__ outA, __bf16* __restrict__ outB,
                            float* __restrict__ outF) {
  constexpr int NT1 = H / 16;
  constexpr int KB2 = H / 32;
  constexpr int NT2 = N2 / 16;
  __shared__ __align__(16) __bf16 s_hid[8][16 * H];
  int tid = threadIdx.x, lane = tid & 31, w = tid >> 5;
  int m = lane & 15, hi = (lane >> 4) << 3;
  int gw = blockIdx.x * (blockDim.x >> 5) + w;
  int nw = gridDim.x * (blockDim.x >> 5);
  int KB1 = (K + 31) >> 5;

  for (int tile = gw; tile < Mtiles; tile += nw) {
    const __bf16* W1 = launder(W1g);
    const __bf16* W2 = launder(W2g);
    int row = tile * 16 + m;
    const float*  rf = nullptr;
    const __bf16* rb = nullptr;
    if constexpr (IN_F32) rf = (const float*)inv + (size_t)row * ldin;
    else                  rb = (const __bf16*)inv + (size_t)row * ldin;

    // layer 1 -> LDS (relu)
    {
      v8f acc[NT1] = {};
#pragma unroll 1
      for (int kb = 0; kb < KB1; ++kb) {
        v16bf a;
        if constexpr (IN_F32) a = afrag_f32(rf, kb * 32, hi, K);
        else                  a = afrag_bf16(rb, kb * 32, hi);
#pragma unroll
        for (int nt = 0; nt < NT1; ++nt)
          acc[nt] = wmma_bf16(a, bfrag(W1, kb * NT1 + nt, lane), acc[nt]);
      }
#pragma unroll
      for (int nt = 0; nt < NT1; ++nt) {
        float bias = b1[nt * 16 + m];
#pragma unroll
        for (int r = 0; r < 8; ++r) {
          float v = acc[nt][r] + bias;
          v = v > 0.f ? v : 0.f;
          s_hid[w][(r + hi) * H + nt * 16 + m] = (__bf16)v;
        }
      }
    }
    wait_lds();

    // layer 2 -> global
    {
      v8f acc[NT2] = {};
#pragma unroll
      for (int kb = 0; kb < KB2; ++kb) {
        v16bf a = afrag_bf16(&s_hid[w][m * H], kb * 32, hi);
#pragma unroll
        for (int nt = 0; nt < NT2; ++nt)
          acc[nt] = wmma_bf16(a, bfrag(W2, kb * NT2 + nt, lane), acc[nt]);
      }
#pragma unroll
      for (int nt = 0; nt < NT2; ++nt) {
        float bias = b2[nt * 16 + m];
#pragma unroll
        for (int r = 0; r < 8; ++r) {
          float v = acc[nt][r] + bias;
          if (RELU2) v = v > 0.f ? v : 0.f;
          size_t o = (size_t)(tile * 16 + r + hi) * N2 + nt * 16 + m;
          if (outF) outF[o] = v;
          __bf16 bv = (__bf16)v;
          if (outA) outA[o] = bv;
          if (outB) outB[o] = bv;
        }
      }
    }
  }
}

// ---------------------------------------------------------------------------
// One MPN step, fully fused. One wave per 16-edge tile:
//   h   = relu([nf0[i]|nf[i]|nf0[j]|nf[j]|ef0|ef] @ Wm0 + bm0)   (768->64)
//   ef' = relu(h @ Wm1 + bm1)                                     (64->128)
//   msg = relu([nf0[i]|nf[i]|ef'] @ Wn0 + bn0)                    (384->128)
//   nf_acc[i] += msg   (atomic f32 scatter = segment_sum)
// ef updated in place (each tile owns its 16 rows). Concats are free:
// 128-wide blocks align with K-blocks of 32 -> pure pointer selection.
// ---------------------------------------------------------------------------
__global__ void mpn_step_kernel(const int* __restrict__ eidx,
                                const __bf16* __restrict__ nf0b, const __bf16* __restrict__ nfb,
                                const __bf16* __restrict__ ef0b, __bf16* __restrict__ efb,
                                const __bf16* __restrict__ Wm0g, const float* __restrict__ bm0,
                                const __bf16* __restrict__ Wm1g, const float* __restrict__ bm1,
                                const __bf16* __restrict__ Wn0g, const float* __restrict__ bn0,
                                float* __restrict__ nf_acc, int E) {
  __shared__ __align__(16) __bf16 s_h[8][16 * 64];
  __shared__ __align__(16) __bf16 s_ef[8][16 * 128];
  __shared__ int s_i[8][16];
  int tid = threadIdx.x, lane = tid & 31, w = tid >> 5;
  int m = lane & 15, hi = (lane >> 4) << 3;
  int gw = blockIdx.x * (blockDim.x >> 5) + w;
  int nw = gridDim.x * (blockDim.x >> 5);
  int tiles = E >> 4;

  for (int tile = gw; tile < tiles; tile += nw) {
    const __bf16* Wm0 = launder(Wm0g);
    const __bf16* Wm1 = launder(Wm1g);
    const __bf16* Wn0 = launder(Wn0g);
    int e0 = tile << 4;
    int jj = eidx[e0 + m];       // source j = edge_index[0]
    int ii = eidx[E + e0 + m];   // target i = edge_index[1]
    if (lane < 16) s_i[w][lane] = ii;
    const __bf16* p0 = nf0b + (size_t)ii * FD;
    const __bf16* p1 = nfb  + (size_t)ii * FD;
    const __bf16* p2 = nf0b + (size_t)jj * FD;
    const __bf16* p3 = nfb  + (size_t)jj * FD;
    const __bf16* p4 = ef0b + (size_t)(e0 + m) * FD;
    const __bf16* p5 = efb  + (size_t)(e0 + m) * FD;

    // ---- stage 1: h [16x64] = relu(concat @ Wm0 + bm0), K=768 ----
    {
      v8f acc[4] = {};
      auto mac4 = [&](const __bf16* rp, int kbBase) {
#pragma unroll
        for (int kk = 0; kk < 4; ++kk) {
          v16bf a = afrag_bf16(rp, kk * 32, hi);
#pragma unroll
          for (int nt = 0; nt < 4; ++nt)
            acc[nt] = wmma_bf16(a, bfrag(Wm0, (kbBase + kk) * 4 + nt, lane), acc[nt]);
        }
      };
      mac4(p0, 0); mac4(p1, 4); mac4(p2, 8);
      mac4(p3, 12); mac4(p4, 16); mac4(p5, 20);
#pragma unroll
      for (int nt = 0; nt < 4; ++nt) {
        float bias = bm0[nt * 16 + m];
#pragma unroll
        for (int r = 0; r < 8; ++r) {
          float v = acc[nt][r] + bias;
          v = v > 0.f ? v : 0.f;
          s_h[w][(r + hi) * 64 + nt * 16 + m] = (__bf16)v;
        }
      }
    }
    wait_lds();

    // ---- stage 2: ef' [16x128] = relu(h @ Wm1 + bm1), K=64 ----
    {
      v8f acc[8] = {};
#pragma unroll
      for (int kb = 0; kb < 2; ++kb) {
        v16bf a = afrag_bf16(&s_h[w][m * 64], kb << 5, hi);
#pragma unroll
        for (int nt = 0; nt < 8; ++nt)
          acc[nt] = wmma_bf16(a, bfrag(Wm1, kb * 8 + nt, lane), acc[nt]);
      }
#pragma unroll
      for (int nt = 0; nt < 8; ++nt) {
        float bias = bm1[nt * 16 + m];
#pragma unroll
        for (int r = 0; r < 8; ++r) {
          float v = acc[nt][r] + bias;
          v = v > 0.f ? v : 0.f;
          s_ef[w][(r + hi) * 128 + nt * 16 + m] = (__bf16)v;
        }
      }
    }
    wait_lds();

    // cooperative vectorized writeback of ef tile (in place): 128B per lane
    {
      int row = lane >> 1, half = lane & 1;
#pragma unroll
      for (int c = 0; c < 8; ++c) {
        *(v8bf*)(efb + (size_t)(e0 + row) * FD + half * 64 + c * 8) =
            *(const v8bf*)(&s_ef[w][row * 128 + half * 64 + c * 8]);
      }
    }

    // ---- stage 3: msg [16x128] = relu([x_i|ef'] @ Wn0 + bn0), K=384 ----
    {
      v8f acc[8] = {};
      auto mac4n = [&](const __bf16* rp, int kbBase) {
#pragma unroll
        for (int kk = 0; kk < 4; ++kk) {
          v16bf a = afrag_bf16(rp, kk * 32, hi);
#pragma unroll
          for (int nt = 0; nt < 8; ++nt)
            acc[nt] = wmma_bf16(a, bfrag(Wn0, (kbBase + kk) * 8 + nt, lane), acc[nt]);
        }
      };
      mac4n(p0, 0); mac4n(p1, 4); mac4n(&s_ef[w][m * 128], 8);
#pragma unroll
      for (int nt = 0; nt < 8; ++nt) {
        float bias = bn0[nt * 16 + m];
#pragma unroll
        for (int r = 0; r < 8; ++r) {
          float v = acc[nt][r] + bias;
          v = v > 0.f ? v : 0.f;
          int irow = s_i[w][r + hi];
          atomAddF(&nf_acc[(size_t)irow * FD + nt * 16 + m], v);
        }
      }
    }
  }
}

// classifier last layer: out[e] = h2[e,:] . cw2 + cb2   (32->1, trivial)
__global__ void final_dot_kernel(const float* __restrict__ h2, const float* __restrict__ wv,
                                 const float* __restrict__ bv, float* __restrict__ out, int E) {
  int e = blockIdx.x * blockDim.x + threadIdx.x;
  if (e >= E) return;
  float s = bv[0];
#pragma unroll
  for (int k = 0; k < 32; ++k) s += h2[(size_t)e * 32 + k] * wv[k];
  out[e] = s;
}

// ---------------------------------------------------------------------------
extern "C" void kernel_launch(void* const* d_in, const int* in_sizes, int n_in,
                              void* d_out, int out_size, void* d_ws, size_t ws_size,
                              hipStream_t stream) {
  const float* x    = (const float*)d_in[0];
  const float* ea   = (const float*)d_in[1];
  const int*   ei   = (const int*)d_in[2];
  const float* new0 = (const float*)d_in[3];  const float* neb0 = (const float*)d_in[4];
  const float* new1 = (const float*)d_in[5];  const float* neb1 = (const float*)d_in[6];
  const float* eew0 = (const float*)d_in[7];  const float* eeb0 = (const float*)d_in[8];
  const float* eew1 = (const float*)d_in[9];  const float* eeb1 = (const float*)d_in[10];
  const float* mew0 = (const float*)d_in[11]; const float* meb0 = (const float*)d_in[12];
  const float* mew1 = (const float*)d_in[13]; const float* meb1 = (const float*)d_in[14];
  const float* mnw0 = (const float*)d_in[15]; const float* mnb0 = (const float*)d_in[16];
  const float* cw0  = (const float*)d_in[17]; const float* cb0  = (const float*)d_in[18];
  const float* cw1  = (const float*)d_in[19]; const float* cb1  = (const float*)d_in[20];
  const float* cw2  = (const float*)d_in[21]; const float* cb2  = (const float*)d_in[22];

  char* ws = (char*)d_ws;
  size_t off = 0;
  auto alloc = [&](size_t bytes) -> char* {
    off = (off + 255) & ~(size_t)255;
    char* p = ws + off;
    off += bytes;
    return p;
  };
  auto psz = [](int K, int N) -> size_t {
    return (size_t)((K + 31) / 32) * ((N + 15) / 16) * 512 * sizeof(__bf16);
  };

  __bf16* new0p = (__bf16*)alloc(psz(128, 128));
  __bf16* new1p = (__bf16*)alloc(psz(128, 128));
  __bf16* eew0p = (__bf16*)alloc(psz(291, 64));
  __bf16* eew1p = (__bf16*)alloc(psz(64, 128));
  __bf16* mew0p = (__bf16*)alloc(psz(768, 64));
  __bf16* mew1p = (__bf16*)alloc(psz(64, 128));
  __bf16* mnw0p = (__bf16*)alloc(psz(384, 128));
  __bf16* cw0p  = (__bf16*)alloc(psz(128, 64));
  __bf16* cw1p  = (__bf16*)alloc(psz(64, 32));

  __bf16* nf0b   = (__bf16*)alloc((size_t)NNODE * FD * sizeof(__bf16));
  __bf16* nfb    = (__bf16*)alloc((size_t)NNODE * FD * sizeof(__bf16));
  __bf16* ef0b   = (__bf16*)alloc((size_t)NEDGE * FD * sizeof(__bf16));
  __bf16* efb    = (__bf16*)alloc((size_t)NEDGE * FD * sizeof(__bf16));
  float*  nf_acc = (float*) alloc((size_t)NNODE * FD * sizeof(float));
  float*  h2buf  = (float*) alloc((size_t)NEDGE * 32 * sizeof(float));

  auto pack = [&](const float* W, __bf16* dst, int K, int N) {
    int KB = (K + 31) / 32, NT = (N + 15) / 16;
    int total = KB * NT * 512;
    pack_w_kernel<<<(total + 255) / 256, 256, 0, stream>>>(W, dst, K, N, KB, NT);
  };
  pack(new0, new0p, 128, 128);
  pack(new1, new1p, 128, 128);
  pack(eew0, eew0p, 291, 64);
  pack(eew1, eew1p, 64, 128);
  pack(mew0, mew0p, 768, 64);
  pack(mew1, mew1p, 64, 128);
  pack(mnw0, mnw0p, 384, 128);
  pack(cw0,  cw0p,  128, 64);
  pack(cw1,  cw1p,  64, 32);

  const int BLK = 256;                 // 8 waves (wave32)
  const int WPB = BLK / 32;

  // node embedding: x(f32,128) -> 128 -> 128, no final relu; dual bf16 output
  {
    int tiles = NNODE / 16;
    mlp2_kernel<true, false, 128, 128><<<(tiles + WPB - 1) / WPB, BLK, 0, stream>>>(
        x, 128, tiles, 128, new0p, neb0, new1p, neb1,
        nf0b, nfb, nullptr);
  }
  // edge embedding: edge_attr(f32,291) -> 64 -> 128, no final relu; dual bf16 out
  {
    int tiles = NEDGE / 16;
    mlp2_kernel<true, false, 64, 128><<<(tiles + WPB - 1) / WPB, BLK, 0, stream>>>(
        ea, 291, tiles, 291, eew0p, eeb0, eew1p, eeb1,
        ef0b, efb, nullptr);
  }

  // 4 message-passing steps
  {
    int tiles = NEDGE / 16;
    int blocks = (tiles + WPB - 1) / WPB;
    int nElems = NNODE * FD;
    for (int s = 0; s < 4; ++s) {
      zero_f32_kernel<<<(nElems + 255) / 256, 256, 0, stream>>>(nf_acc, nElems);
      mpn_step_kernel<<<blocks, BLK, 0, stream>>>(
          ei, nf0b, nfb, ef0b, efb,
          mew0p, meb0, mew1p, meb1, mnw0p, mnb0,
          nf_acc, NEDGE);
      cvt_f32_bf16_kernel<<<(nElems + 255) / 256, 256, 0, stream>>>(nf_acc, nfb, nElems);
    }
  }

  // classifier head: ef(bf16,128) -> 64 -> 32 (relu,relu), f32 out; then 32->1
  {
    int tiles = NEDGE / 16;
    mlp2_kernel<false, true, 64, 32><<<(tiles + WPB - 1) / WPB, BLK, 0, stream>>>(
        efb, 128, tiles, 128, cw0p, cb0, cw1p, cb1,
        nullptr, nullptr, h2buf);
    final_dot_kernel<<<(NEDGE + 255) / 256, 256, 0, stream>>>(
        h2buf, cw2, cb2, (float*)d_out, NEDGE);
  }
}
